// HTGNNLayer_1786706395357
// MI455X (gfx1250) — matrix operand order repaired and checked by reading, not compiled, not run on hardware
//
#include <hip/hip_runtime.h>
#include <hip/hip_bf16.h>
#include <math.h>

// Problem constants (match reference)
#define NN 40000
#define DD 128
#define TT 6
#define EE 256000
#define RR 4
#define EPSLN 1e-5f

typedef __attribute__((ext_vector_type(16))) _Float16 v16h;
typedef __attribute__((ext_vector_type(8)))  float    v8f;

// Workspace float counts
#define AGG_CNT ((size_t)RR * TT * NN * DD)   // 122,880,000
#define DEG_CNT ((size_t)RR * TT * NN)        // 960,000
#define ATT_CNT ((size_t)RR * TT)             // 24
#define GI_CNT  ((size_t)RR * TT * NN * 3)    // 2,880,000

// ---------------------------------------------------------------- zero fill
__global__ void htg_zero_kernel(float* __restrict__ p, size_t count) {
    size_t i = (size_t)blockIdx.x * blockDim.x + threadIdx.x;
    size_t stride = (size_t)gridDim.x * blockDim.x;
    for (; i < count; i += stride) p[i] = 0.0f;
}

// ------------------------------------------------- edge scatter (wave/edge)
// one wave32 per edge; each lane moves 4 contiguous floats (float4 gather)
__global__ void htg_scatter_kernel(const float* __restrict__ feat_a,
                                   const float* __restrict__ feat_b,
                                   const int*   __restrict__ src_idx,
                                   const int*   __restrict__ dst_idx,
                                   float* __restrict__ agg,
                                   float* __restrict__ deg) {
    const int wid  = (blockIdx.x * blockDim.x + threadIdx.x) >> 5;
    const int lane = threadIdx.x & 31;
    const int totalEdges = RR * TT * EE;              // 6,144,000
    if (wid >= totalEdges) return;
    const int r   = wid / (TT * EE);
    const int rem = wid - r * (TT * EE);
    const int t   = rem / EE;
    const int e   = rem - t * EE;

    const size_t ib = (((size_t)r * TT + t) * EE) + e;
    const int s   = src_idx[ib];
    const int dst = dst_idx[ib];

    const float* feat = (r & 1) ? feat_b : feat_a;    // SRC_OF = [0,1,0,1]
    const float4 v = *((const float4*)(feat + ((size_t)t * NN + s) * DD) + lane);

    float* dp = agg + (((size_t)r * TT + t) * NN + dst) * DD + lane * 4;
    atomicAdd(dp + 0, v.x);
    atomicAdd(dp + 1, v.y);
    atomicAdd(dp + 2, v.z);
    atomicAdd(dp + 3, v.w);
    if (lane == 0) atomicAdd(deg + ((size_t)r * TT + t) * NN + dst, 1.0f);
}

// ------------------------------------------- degree normalize + ELU inplace
__global__ void htg_norm_elu_kernel(float* __restrict__ agg,
                                    const float* __restrict__ deg) {
    const size_t total4 = (size_t)RR * TT * NN * (DD / 4);
    size_t i = (size_t)blockIdx.x * blockDim.x + threadIdx.x;
    const size_t stride = (size_t)gridDim.x * blockDim.x;
    float4* p = (float4*)agg;
    for (; i < total4; i += stride) {
        const size_t node = i / (DD / 4);
        const float inv = 1.0f / fmaxf(deg[node], 1.0f);
        float4 v = p[i];
        v.x *= inv; v.y *= inv; v.z *= inv; v.w *= inv;
        v.x = (v.x > 0.0f) ? v.x : expm1f(v.x);
        v.y = (v.y > 0.0f) ? v.y : expm1f(v.y);
        v.z = (v.z > 0.0f) ? v.z : expm1f(v.z);
        v.w = (v.w > 0.0f) ? v.w : expm1f(v.w);
        p[i] = v;
    }
}

// -------------------------------------------------- WMMA: gi = x @ w_ih^T + b_ih
// One wave per 16-node tile of one (r,t).  A = x tile [16 x 128] (f16),
// B = w_ih^T K-chunk [32 x 16] with only columns 0..2 nonzero, f32 accum.
// ISA 16-bit A 16x32 layout: lane<16 row M=lane holds K {0..7,16..23};
// lane>=16 holds K {8..15,24..31}  -> two contiguous 8-float runs per lane.
// B 32x16: half-wave K split, col = lane&15 -> two contiguous 8-float runs, K order as A.
// C/D: col = lane&15, VGPR v = row (v + 8*(lane>>4)).
// B is masked (multiply by 0/1), never branch-guarded: EXEC stays all-1s and
// loads stay wide (global_load_b128), no saveexec chains.
__global__ void htg_gi_wmma_kernel(const float* __restrict__ rel_out,
                                   const float* __restrict__ w_ih,
                                   const float* __restrict__ b_ih,
                                   float* __restrict__ gi) {
    const int wave = (blockIdx.x * blockDim.x + threadIdx.x) >> 5;
    const int lane = threadIdx.x & 31;
    const int tilesPerRT = NN / 16;                   // 2500
    const int totalTiles = RR * TT * tilesPerRT;      // 60000
    if (wave >= totalTiles) return;                   // wave-uniform guard

    const int rt   = wave / tilesPerRT;
    const int tile = wave - rt * tilesPerRT;
    const int r    = rt / TT;

    const int half = lane >> 4;                       // 0 or 1
    const int lrow = lane & 15;

    const float* xrow  = rel_out + ((size_t)rt * NN + (size_t)tile * 16 + lrow) * DD;
    const float* wrow  = w_ih + ((size_t)r * 3 + (lrow < 3 ? lrow : 0)) * DD;
    const float  bmask = (lrow < 3) ? 1.0f : 0.0f;    // zero-pad cols 3..15

    v8f acc = {};
#pragma unroll
    for (int c = 0; c < 4; ++c) {                     // K = 128 in chunks of 32
        // ---- A fragment: two contiguous 8-float runs -> 4x float4 loads
        const float4* ap0 = (const float4*)(xrow + c * 32 + half * 8);
        const float4* ap1 = (const float4*)(xrow + c * 32 + 16 + half * 8);
        const float4 a0 = ap0[0], a1 = ap0[1];        // K run {0..7}(+half*8)
        const float4 a2 = ap1[0], a3 = ap1[1];        // K run {16..23}(+half*8)
        v16h a;
        a[0]  = (_Float16)a0.x;  a[1]  = (_Float16)a0.y;
        a[2]  = (_Float16)a0.z;  a[3]  = (_Float16)a0.w;
        a[4]  = (_Float16)a1.x;  a[5]  = (_Float16)a1.y;
        a[6]  = (_Float16)a1.z;  a[7]  = (_Float16)a1.w;
        a[8]  = (_Float16)a2.x;  a[9]  = (_Float16)a2.y;
        a[10] = (_Float16)a2.z;  a[11] = (_Float16)a2.w;
        a[12] = (_Float16)a3.x;  a[13] = (_Float16)a3.y;
        a[14] = (_Float16)a3.z;  a[15] = (_Float16)a3.w;

        // ---- B fragment: two contiguous 8-float runs (same K order as A) -> 4x float4 loads
        const float4* bp0 = (const float4*)(wrow + c * 32 + half * 8);
        const float4* bp1 = (const float4*)(wrow + c * 32 + 16 + half * 8);
        const float4 b0 = bp0[0], b1 = bp0[1], b2 = bp1[0], b3 = bp1[1];
        v16h b;
        b[0]  = (_Float16)(b0.x * bmask);  b[1]  = (_Float16)(b0.y * bmask);
        b[2]  = (_Float16)(b0.z * bmask);  b[3]  = (_Float16)(b0.w * bmask);
        b[4]  = (_Float16)(b1.x * bmask);  b[5]  = (_Float16)(b1.y * bmask);
        b[6]  = (_Float16)(b1.z * bmask);  b[7]  = (_Float16)(b1.w * bmask);
        b[8]  = (_Float16)(b2.x * bmask);  b[9]  = (_Float16)(b2.y * bmask);
        b[10] = (_Float16)(b2.z * bmask);  b[11] = (_Float16)(b2.w * bmask);
        b[12] = (_Float16)(b3.x * bmask);  b[13] = (_Float16)(b3.y * bmask);
        b[14] = (_Float16)(b3.z * bmask);  b[15] = (_Float16)(b3.w * bmask);

        acc = __builtin_amdgcn_wmma_f32_16x16x32_f16(
            /*neg_a=*/false, a, /*neg_b=*/false, b,
            /*c_mod=*/(short)0, acc, /*reuse_a=*/false, /*reuse_b=*/false);
    }

    if (lrow < 3) {
        const float bias = b_ih[r * 3 + lrow];
#pragma unroll
        for (int v = 0; v < 8; ++v) {
            const int node = tile * 16 + half * 8 + v;
            gi[((size_t)rt * NN + node) * 3 + lrow] = acc[v] + bias;
        }
    }
}

// ---------------------------------------- per-node scalar GRU + mean over N
__global__ void htg_gru_att_kernel(const float* __restrict__ gi,
                                   const float* __restrict__ w_hh,
                                   const float* __restrict__ b_hh,
                                   const float* __restrict__ init_att,
                                   float* __restrict__ att_sum) {
    const int r = blockIdx.y;
    const int n = blockIdx.x * blockDim.x + threadIdx.x;
    const bool active = (n < NN);

    const float w0 = w_hh[r * 3 + 0], w1 = w_hh[r * 3 + 1], w2 = w_hh[r * 3 + 2];
    const float c0 = b_hh[r * 3 + 0], c1 = b_hh[r * 3 + 1], c2 = b_hh[r * 3 + 2];

    float h = active ? init_att[r] : 0.0f;
    float outs[TT];
#pragma unroll
    for (int t = 0; t < TT; ++t) {
        float g0 = 0.0f, g1 = 0.0f, g2 = 0.0f;
        if (active) {
            const float* gp = gi + (((size_t)r * TT + t) * NN + n) * 3;
            g0 = gp[0]; g1 = gp[1]; g2 = gp[2];
        }
        const float gh0 = h * w0 + c0;
        const float gh1 = h * w1 + c1;
        const float gh2 = h * w2 + c2;
        const float rg = 1.0f / (1.0f + expf(-(g0 + gh0)));
        const float z  = 1.0f / (1.0f + expf(-(g1 + gh1)));
        const float ng = tanhf(g2 + rg * gh2);
        const float hn = (1.0f - z) * ng + z * h;
        h = hn;
        outs[t] = active ? hn : 0.0f;
    }

    const int lane = threadIdx.x & 31;
#pragma unroll
    for (int t = 0; t < TT; ++t) {
        float v = outs[t];
#pragma unroll
        for (int off = 16; off > 0; off >>= 1) v += __shfl_down(v, off, 32);
        if (lane == 0) atomicAdd(&att_sum[r * TT + t], v);
    }
}

// ------------------------- fuse: softmax(att) weighted sum + LayerNorm(D)
// one wave32 per output row of 128 floats
__global__ void htg_fuse_kernel(const float* __restrict__ rel_out,
                                const float* __restrict__ att_sum,
                                const float* __restrict__ gamma,
                                const float* __restrict__ beta,
                                float* __restrict__ out) {
    const int wid  = (blockIdx.x * blockDim.x + threadIdx.x) >> 5;
    const int lane = threadIdx.x & 31;
    const int totalRows = 2 * TT * NN;                // 480,000
    if (wid >= totalRows) return;
    const int nt  = wid / (TT * NN);
    const int rem = wid - nt * (TT * NN);
    const int t   = rem / NN;
    const int n   = rem - t * NN;
    const int r0  = nt * 2, r1 = nt * 2 + 1;

    const float a0 = att_sum[r0 * TT + t] * (1.0f / (float)NN);
    const float a1 = att_sum[r1 * TT + t] * (1.0f / (float)NN);
    const float m  = fmaxf(a0, a1);
    const float e0 = expf(a0 - m), e1 = expf(a1 - m);
    const float inv = 1.0f / (e0 + e1);
    const float w0 = e0 * inv, w1 = e1 * inv;

    const float4 f0 = *((const float4*)(rel_out + (((size_t)r0 * TT + t) * NN + n) * DD) + lane);
    const float4 f1 = *((const float4*)(rel_out + (((size_t)r1 * TT + t) * NN + n) * DD) + lane);
    float4 f;
    f.x = w0 * f0.x + w1 * f1.x;
    f.y = w0 * f0.y + w1 * f1.y;
    f.z = w0 * f0.z + w1 * f1.z;
    f.w = w0 * f0.w + w1 * f1.w;

    float s  = f.x + f.y + f.z + f.w;
    float sq = f.x * f.x + f.y * f.y + f.z * f.z + f.w * f.w;
#pragma unroll
    for (int off = 16; off > 0; off >>= 1) {
        s  += __shfl_xor(s,  off, 32);
        sq += __shfl_xor(sq, off, 32);
    }
    const float mu   = s * (1.0f / (float)DD);
    const float var  = sq * (1.0f / (float)DD) - mu * mu;
    const float rstd = rsqrtf(var + EPSLN);

    const float* g = gamma + (size_t)nt * DD + lane * 4;
    const float* b = beta  + (size_t)nt * DD + lane * 4;
    float4 o;
    o.x = (f.x - mu) * rstd * g[0] + b[0];
    o.y = (f.y - mu) * rstd * g[1] + b[1];
    o.z = (f.z - mu) * rstd * g[2] + b[2];
    o.w = (f.w - mu) * rstd * g[3] + b[3];
    *((float4*)(out + (((size_t)nt * TT + t) * NN + n) * DD) + lane) = o;
}

// ---------------------------------------------------------------------------
extern "C" void kernel_launch(void* const* d_in, const int* in_sizes, int n_in,
                              void* d_out, int out_size, void* d_ws, size_t ws_size,
                              hipStream_t stream) {
    (void)in_sizes; (void)n_in; (void)out_size; (void)ws_size;

    const float* feat_a   = (const float*)d_in[0];
    const float* feat_b   = (const float*)d_in[1];
    const int*   src_idx  = (const int*)  d_in[2];
    const int*   dst_idx  = (const int*)  d_in[3];
    const float* w_ih     = (const float*)d_in[4];
    const float* w_hh     = (const float*)d_in[5];
    const float* b_ih     = (const float*)d_in[6];
    const float* b_hh     = (const float*)d_in[7];
    const float* init_att = (const float*)d_in[8];
    const float* gamma    = (const float*)d_in[9];
    const float* beta     = (const float*)d_in[10];
    float* out = (float*)d_out;

    float* ws      = (float*)d_ws;
    float* agg     = ws;                              // becomes rel_out in place
    float* deg     = agg + AGG_CNT;
    float* att_sum = deg + DEG_CNT;
    float* gi      = att_sum + ATT_CNT;

    const int BLK = 256;

    // 1) zero agg + deg + att_sum (contiguous)
    htg_zero_kernel<<<8192, BLK, 0, stream>>>(agg, AGG_CNT + DEG_CNT + ATT_CNT);

    // 2) edge scatter: one wave per edge -> 6,144,000 waves, 8 per block
    htg_scatter_kernel<<<(RR * TT * EE) / 8, BLK, 0, stream>>>(
        feat_a, feat_b, src_idx, dst_idx, agg, deg);

    // 3) degree-normalize + ELU in place
    htg_norm_elu_kernel<<<8192, BLK, 0, stream>>>(agg, deg);

    // 4) WMMA gate pre-activations gi[R,T,N,3]
    htg_gi_wmma_kernel<<<(RR * TT * (NN / 16)) / 8, BLK, 0, stream>>>(
        agg, w_ih, b_ih, gi);

    // 5) per-node scalar GRU scan + mean over nodes
    dim3 ggrid((NN + BLK - 1) / BLK, RR);
    htg_gru_att_kernel<<<ggrid, BLK, 0, stream>>>(gi, w_hh, b_hh, init_att, att_sum);

    // 6) softmax-weighted fuse + LayerNorm -> out[2,T,N,D]
    htg_fuse_kernel<<<(2 * TT * NN) / 8, BLK, 0, stream>>>(
        agg, att_sum, gamma, beta, out);
}
